// DynamicGraphBuilder_3573412790306
// MI455X (gfx1250) — matrix operand order repaired and checked by reading, not compiled, not run on hardware
//
#include <hip/hip_runtime.h>

typedef __attribute__((ext_vector_type(16))) _Float16 v16h;
typedef __attribute__((ext_vector_type(8)))  _Float16 v8h;
typedef __attribute__((ext_vector_type(4)))  _Float16 v4h;
typedef __attribute__((ext_vector_type(8)))  float    v8f;

#define BATCH 8
#define NPTS  2048
#define DIM   128
#define TOPK  8
#define ROWS  16                      // rows per workgroup (one WMMA M tile)
#define WAVES 4
#define COLS_PER_WAVE (NPTS / WAVES)  // 512
#define SUB   128                     // columns scored per LDS chunk
#define NSLOT 8                       // running top-8 slots per row
#define ROWLEN (SUB + NSLOT + 1)      // 137: pad so row stride dodges 64-bank conflicts
#define NEG_INF (-3.402823466e38f)

union AFrag { v16h v; v8h h[2]; };

// ---------------------------------------------------------------------------
// 1) L2-normalize rows, emit f16. One wave32 per row of 128 features.
// ---------------------------------------------------------------------------
__global__ __launch_bounds__(256) void normalize_kernel(const float* __restrict__ x,
                                                        _Float16* __restrict__ out)
{
    const int gid  = blockIdx.x * blockDim.x + threadIdx.x;
    const int rowi = gid >> 5;       // 32 lanes x 4 floats = 128 = DIM
    const int lane = gid & 31;

    const float4 v = ((const float4*)(x + (size_t)rowi * DIM))[lane];
    float s = v.x * v.x + v.y * v.y + v.z * v.z + v.w * v.w;
#pragma unroll
    for (int off = 16; off > 0; off >>= 1) s += __shfl_xor(s, off);

    const float scale = 1.0f / fmaxf(sqrtf(s), 1e-12f);
    v4h o;
    o[0] = (_Float16)(v.x * scale);
    o[1] = (_Float16)(v.y * scale);
    o[2] = (_Float16)(v.z * scale);
    o[3] = (_Float16)(v.w * scale);
    ((v4h*)(out + (size_t)rowi * DIM))[lane] = o;
}

// ---------------------------------------------------------------------------
// 2) Cosine scores via v_wmma_f32_16x16x32_f16 + fused per-row top-8.
//    Block = 4 waves; each wave scores a 16x512 strip in 16x128 LDS chunks.
// ---------------------------------------------------------------------------
__global__ __launch_bounds__(WAVES * 32) void score_topk_kernel(
    const _Float16* __restrict__ normed, int* __restrict__ idxOut)
{
    __shared__ float sc[WAVES][ROWS][ROWLEN];          // 35 KB
    __shared__ int   slotIdx[WAVES][ROWS][NSLOT];      // 2 KB
    __shared__ float mVal[ROWS][WAVES * NSLOT];        // 2 KB
    __shared__ int   mIdx[ROWS][WAVES * NSLOT];        // 2 KB

    const int b    = blockIdx.y;
    const int m0   = blockIdx.x * ROWS;
    const int tid  = threadIdx.x;
    const int wave = tid >> 5;
    const int lane = tid & 31;
    const int l15  = lane & 15;
    const int half = lane >> 4;

    const _Float16* base = normed + (size_t)b * NPTS * DIM;

    // A fragments (16x32 f16, M=m0..m0+15), resident for the whole kernel.
    // Layout: lane row M=l15 holds K contiguous: K=h*16+0..15 (matches B fragment).
    const _Float16* arow = base + (size_t)(m0 + l15) * DIM;
    AFrag afrag[4];
#pragma unroll
    for (int kt = 0; kt < 4; ++kt) {
        afrag[kt].h[0] = *(const v8h*)(arow + kt * 32 + half * 16);
        afrag[kt].h[1] = *(const v8h*)(arow + kt * 32 + half * 16 + 8);
    }

    if (lane < NSLOT)
        for (int r = 0; r < ROWS; ++r) {
            sc[wave][r][SUB + lane] = NEG_INF;
            slotIdx[wave][r][lane]  = 0;
        }

    for (int sub = 0; sub < COLS_PER_WAVE / SUB; ++sub) {
        const int colbase = wave * COLS_PER_WAVE + sub * SUB;

        // ---- score 8 column tiles (16x16 each) into the LDS chunk ----
        for (int ct = 0; ct < SUB / 16; ++ct) {
            const int n0 = colbase + ct * 16;
            // B fragment 32x16: lane = column n (= row n of normed),
            // K contiguous: lanes 0-15 K=0..15, lanes 16-31 K=16..31.
            const _Float16* brow = base + (size_t)(n0 + l15) * DIM + half * 16;
            v8f acc = {};
#pragma unroll
            for (int kt = 0; kt < 4; ++kt) {
                v16h bfrag = *(const v16h*)(brow + kt * 32);
                acc = __builtin_amdgcn_wmma_f32_16x16x32_f16(
                          false, afrag[kt].v, false, bfrag,
                          (short)0, acc, false, false);
            }
            // D fragment: lane -> col l15, VGPR r -> row half*8+r
#pragma unroll
            for (int r = 0; r < 8; ++r)
                sc[wave][half * 8 + r][ct * 16 + l15] = acc[r];
        }
        asm volatile("s_wait_dscnt 0" ::: "memory");
        __builtin_amdgcn_wave_barrier();

        // ---- per-row top-8 over {128 fresh scores} U {8 running slots} ----
        for (int r = 0; r < ROWS; ++r) {
            float* row = &sc[wave][r][0];
            float myNewVal = NEG_INF;
            int   myNewIdx = 0;
            for (int pass = 0; pass < TOPK; ++pass) {
                float best = NEG_INF; int bi = -1; int gi = 0x7fffffff;
                for (int j = lane; j < SUB + NSLOT; j += 32) {
                    float v = row[j];
                    int   g = (j < SUB) ? (colbase + j) : slotIdx[wave][r][lane & 7];
                    if (v > best) { best = v; bi = j; gi = g; }
                }
#pragma unroll
                for (int off = 16; off > 0; off >>= 1) {
                    float ov  = __shfl_xor(best, off);
                    int   obi = __shfl_xor(bi, off);
                    int   ogi = __shfl_xor(gi, off);
                    if (ov > best || (ov == best && ogi < gi)) { best = ov; bi = obi; gi = ogi; }
                }
                // invalidate winner; writer lane == every future reader lane
                if (bi >= 0 && (bi & 31) == lane) row[bi] = NEG_INF;
                if (lane == pass) { myNewVal = best; myNewIdx = gi; }
            }
            if (lane < NSLOT) {                     // lane s owns slot s (same-lane RW)
                row[SUB + lane]        = myNewVal;
                slotIdx[wave][r][lane] = myNewIdx;
            }
        }
        __builtin_amdgcn_wave_barrier();
    }

    // ---- inter-wave merge: 4 waves x 8 slots = 32 distinct candidates/row ----
    if (lane < NSLOT)
        for (int r = 0; r < ROWS; ++r) {
            mVal[r][wave * NSLOT + lane] = sc[wave][r][SUB + lane];
            mIdx[r][wave * NSLOT + lane] = slotIdx[wave][r][lane];
        }
    __syncthreads();

    for (int r = wave; r < ROWS; r += WAVES) {
        float v = mVal[r][lane];
        int   g = mIdx[r][lane];
        for (int pass = 0; pass < TOPK; ++pass) {
            float best = v; int gi = g; int bl = lane;
#pragma unroll
            for (int off = 16; off > 0; off >>= 1) {
                float ov  = __shfl_xor(best, off);
                int   ogi = __shfl_xor(gi, off);
                int   obl = __shfl_xor(bl, off);
                if (ov > best || (ov == best && obl < bl)) { best = ov; gi = ogi; bl = obl; }
            }
            if (lane == pass)
                idxOut[(((size_t)b * NPTS + m0 + r) << 3) + pass] = gi;
            if (lane == bl) v = NEG_INF;   // winner retires its candidate
        }
    }
}

// ---------------------------------------------------------------------------
// 3) adj = eye (float4 stores; this is the HBM-bandwidth-dominant step)
// ---------------------------------------------------------------------------
__global__ __launch_bounds__(256) void init_adj_kernel(float* __restrict__ adj)
{
    const size_t t  = (size_t)blockIdx.x * blockDim.x + threadIdx.x;
    const size_t e0 = t * 4;
    const int i = (int)((e0 >> 11) & (NPTS - 1));
    const int j = (int)(e0 & (NPTS - 1));
    float4 v;
    v.x = (j + 0 == i) ? 1.0f : 0.0f;
    v.y = (j + 1 == i) ? 1.0f : 0.0f;
    v.z = (j + 2 == i) ? 1.0f : 0.0f;
    v.w = (j + 3 == i) ? 1.0f : 0.0f;
    ((float4*)adj)[t] = v;
}

// ---------------------------------------------------------------------------
// 4) scatter adjacency: adj[b,n,g]=1 and adj[b,g,n]=1 (symmetrize)
// ---------------------------------------------------------------------------
__global__ __launch_bounds__(256) void scatter_kernel(const int* __restrict__ idx,
                                                      float* __restrict__ adj)
{
    const int t   = blockIdx.x * blockDim.x + threadIdx.x;   // < B*N*TOPK
    const int b   = t >> 14;                                 // N*TOPK = 16384
    const int rem = t & 16383;
    const int n   = rem >> 3;
    const int g   = idx[t];
    float* A = adj + (size_t)b * NPTS * NPTS;
    A[(size_t)n * NPTS + g] = 1.0f;
    A[(size_t)g * NPTS + n] = 1.0f;
}

// ---------------------------------------------------------------------------
extern "C" void kernel_launch(void* const* d_in, const int* in_sizes, int n_in,
                              void* d_out, int out_size, void* d_ws, size_t ws_size,
                              hipStream_t stream)
{
    (void)in_sizes; (void)n_in; (void)out_size; (void)ws_size;

    const float* x   = (const float*)d_in[0];
    float*       adj = (float*)d_out;

    _Float16* normed  = (_Float16*)d_ws;                                  // 4 MB
    int*      topkIdx = (int*)((char*)d_ws +
                               (size_t)BATCH * NPTS * DIM * sizeof(_Float16)); // 512 KB

    normalize_kernel<<<(BATCH * NPTS) / 8, 256, 0, stream>>>(x, normed);
    score_topk_kernel<<<dim3(NPTS / ROWS, BATCH), WAVES * 32, 0, stream>>>(normed, topkIdx);
    init_adj_kernel<<<(int)((size_t)BATCH * NPTS * NPTS / 4 / 256), 256, 0, stream>>>(adj);
    scatter_kernel<<<(BATCH * NPTS * TOPK) / 256, 256, 0, stream>>>(topkIdx, adj);
}
